// FigureGroundSegmentation_87419764342806
// MI455X (gfx1250) — matrix operand order repaired and checked by reading, not compiled, not run on hardware
//
#include <hip/hip_runtime.h>
#include <cstdint>

typedef __attribute__((ext_vector_type(16))) __bf16 v16bf;
typedef __attribute__((ext_vector_type(8)))  float  v8f;

#define B_   128
#define N_   4096
#define DIN  256
#define D_   256
#define H_   1024
#define EPSW 1e-8f

// ---- workspace layout (bytes) ----
#define KOFF  ((size_t)0)            // k bf16  [B*N, 256]            268,435,456
#define VOFF  ((size_t)268435456)    // v bf16  [B*N, 256]            268,435,456
#define WOFF  ((size_t)536870912)    // WkT/WvT bf16 [2][256][256]        262,144
#define QOFF  ((size_t)537133056)    // q f32   [B,4,64]                  131,072
#define UOFF  ((size_t)537264128)    // U f32   [B,4,64]                  131,072
#define ZOFF  ((size_t)537395200)    // Z f32   [B,4]                       2,048
#define SOFF  ((size_t)537397248)    // s f32   [B,4,N]                 8,388,608
#define ROFF  ((size_t)545785856)    // rep f32 [B,256]                   131,072

#if defined(__AMDGCN__) && __has_builtin(__builtin_amdgcn_global_load_async_to_lds_b128) && __has_builtin(__builtin_amdgcn_s_wait_asynccnt)
#define FGS_ASYNC 1
__device__ __forceinline__ void asyncLoad128(const void* gsrc, void* ldst) {
  typedef int v4i __attribute__((vector_size(4 * sizeof(int))));
  typedef __attribute__((address_space(1))) v4i* gp_t;
  typedef __attribute__((address_space(3))) v4i* lp_t;
  __builtin_amdgcn_global_load_async_to_lds_b128((gp_t)gsrc, (lp_t)ldst, 0, 0);
}
#else
#define FGS_ASYNC 0
#endif

__device__ __forceinline__ float bf2f(unsigned int hs) { return __uint_as_float(hs << 16); }
__device__ __forceinline__ unsigned short f2bf(float f) {
  unsigned int u = __float_as_uint(f);
  u = (u + 0x7fffu + ((u >> 16) & 1u)) >> 16;
  return (unsigned short)u;
}
__device__ __forceinline__ void unpack8(uint4 p, float* o) {
  o[0] = bf2f(p.x & 0xffffu); o[1] = bf2f(p.x >> 16);
  o[2] = bf2f(p.y & 0xffffu); o[3] = bf2f(p.y >> 16);
  o[4] = bf2f(p.z & 0xffffu); o[5] = bf2f(p.z >> 16);
  o[6] = bf2f(p.w & 0xffffu); o[7] = bf2f(p.w >> 16);
}
__device__ __forceinline__ float blockSum256(float v, float* red) {
  int lane = threadIdx.x & 31, wid = threadIdx.x >> 5;
#pragma unroll
  for (int m = 16; m >= 1; m >>= 1) v += __shfl_xor(v, m, 32);
  __syncthreads();
  if (lane == 0) red[wid] = v;
  __syncthreads();
  return red[0] + red[1] + red[2] + red[3] + red[4] + red[5] + red[6] + red[7];
}

// ---- 0: rep init + weight transpose/convert to bf16 (k pre-scaled by sqrt(4)=2) ----
__global__ __launch_bounds__(256) void fgs_prep(
    const float* __restrict__ Wk, const float* __restrict__ Wv,
    const float* __restrict__ mu, const float* __restrict__ lv,
    const float* __restrict__ eps, unsigned short* __restrict__ wT,
    float* __restrict__ rep) {
  int tid = blockIdx.x * 256 + threadIdx.x;      // 65536 threads
  int kk = tid >> 8, col = tid & 255;
  wT[(size_t)col * 256 + kk]         = f2bf(Wk[(size_t)kk * 256 + col] * 2.0f);
  wT[65536 + (size_t)col * 256 + kk] = f2bf(Wv[(size_t)kk * 256 + col]);
  if (tid < B_ * D_) {
    int d = tid & 255;
    rep[tid] = mu[d] + expf(0.5f * lv[d]) * eps[tid];
  }
}

// ---- 1: fused LayerNorm + (x@Wk, x@Wv) via v_wmma_f32_16x16x32_bf16 ----
// Block tile: 64 rows x 512 cols. 8 waves; each wave: 4 row-tiles x 4 col-tiles.
__global__ __launch_bounds__(256) void fgs_lnproj(
    const float* __restrict__ x, const float* __restrict__ g, const float* __restrict__ bb,
    const unsigned short* __restrict__ wT,
    unsigned short* __restrict__ kb, unsigned short* __restrict__ vb) {
  __shared__ __attribute__((aligned(16))) unsigned short Alds[64][264];
  int w = threadIdx.x >> 5, lane = threadIdx.x & 31;
  size_t rowBase = (size_t)blockIdx.x * 64;

  // LN phase: wave w normalizes rows w*8 .. w*8+7 into LDS as bf16
#pragma unroll
  for (int rr = 0; rr < 8; ++rr) {
    int r = w * 8 + rr;
    size_t gro = (rowBase + r) * DIN;
    float vals[8];
    float s = 0.f, sq = 0.f;
#pragma unroll
    for (int i = 0; i < 8; ++i) {
      float vv = x[gro + lane + i * 32];
      vals[i] = vv; s += vv; sq += vv * vv;
    }
#pragma unroll
    for (int m = 16; m >= 1; m >>= 1) { s += __shfl_xor(s, m, 32); sq += __shfl_xor(sq, m, 32); }
    float mean = s * (1.f / 256.f);
    float var = sq * (1.f / 256.f) - mean * mean;
    float inv = rsqrtf(var + 1e-5f);
#pragma unroll
    for (int i = 0; i < 8; ++i) {
      int idx = lane + i * 32;
      Alds[r][idx] = f2bf((vals[i] - mean) * inv * g[idx] + bb[idx]);
    }
  }
  __syncthreads();

  // GEMM phase
  union AF { v16bf bf; uint4 q[2]; };
  AF afr[4], bfr;
  v8f acc[4][4] = {};
  int row  = lane & 15;
  int kbA  = (lane >> 4) * 8;    // A: half-wave K 8-block (ISA 7.12.2)
  int kbB  = (lane >> 4) * 8;    // B: half-wave K 8-block (same as A)
  int coln = lane & 15;
  for (int kk = 0; kk < 8; ++kk) {
    int K0 = kk * 32;
#pragma unroll
    for (int rt = 0; rt < 4; ++rt) {
      afr[rt].q[0] = *reinterpret_cast<const uint4*>(&Alds[rt * 16 + row][K0 + kbA]);
      afr[rt].q[1] = *reinterpret_cast<const uint4*>(&Alds[rt * 16 + row][K0 + 16 + kbA]);
    }
#pragma unroll
    for (int j = 0; j < 4; ++j) {
      int ct = w * 4 + j;
      int m = ct >> 4;
      int colg = (ct & 15) * 16 + coln;
      const unsigned short* src = wT + ((size_t)m * 256 + colg) * 256 + K0 + kbB;
      bfr.q[0] = *reinterpret_cast<const uint4*>(src);
      bfr.q[1] = *reinterpret_cast<const uint4*>(src + 16);
#pragma unroll
      for (int rt = 0; rt < 4; ++rt) {
        acc[rt][j] = __builtin_amdgcn_wmma_f32_16x16x32_bf16(
            false, afr[rt].bf, false, bfr.bf, (short)0, acc[rt][j], false, false);
      }
    }
  }
  // store D (16x16 f32 layout: VGPR i -> M=(lane>>4)*8+i, N=lane&15)
  int rb = (lane >> 4) * 8;
#pragma unroll
  for (int j = 0; j < 4; ++j) {
    int ct = w * 4 + j;
    int m = ct >> 4;
    int colg = (ct & 15) * 16 + coln;
    unsigned short* dst = (m == 0) ? kb : vb;
#pragma unroll
    for (int rt = 0; rt < 4; ++rt) {
#pragma unroll
      for (int i = 0; i < 8; ++i) {
        size_t rowg = rowBase + rt * 16 + rb + i;
        dst[rowg * 256 + colg] = f2bf(acc[rt][j][i]);
      }
    }
  }
}

// ---- 2: q = LN_slot(rep) @ Wq ; also zero U,Z accumulators ----
__global__ __launch_bounds__(256) void fgs_q(
    const float* __restrict__ rep, const float* __restrict__ Wq,
    const float* __restrict__ g, const float* __restrict__ bb,
    float* __restrict__ q, float* __restrict__ U, float* __restrict__ Z) {
  __shared__ float ys[256];
  __shared__ float red[8];
  int b = blockIdx.x, t = threadIdx.x;
  float v = rep[(size_t)b * 256 + t];
  float s = blockSum256(v, red);
  float sq = blockSum256(v * v, red);
  float mean = s * (1.f / 256.f);
  float var = sq * (1.f / 256.f) - mean * mean;
  ys[t] = (v - mean) * rsqrtf(var + 1e-5f) * g[t] + bb[t];
  __syncthreads();
  float acc = 0.f;
  for (int d = 0; d < 256; ++d) acc += ys[d] * Wq[(size_t)d * 256 + t];
  q[(size_t)b * 256 + t] = acc;
  U[(size_t)b * 256 + t] = 0.f;
  if (t < 4) Z[b * 4 + t] = 0.f;
}

// ---- 3: one-pass sigmoid attention: U += s*v, Z += s ----
// k/v tiles (64 n x 64 dh bf16) staged in LDS; async double-buffered when available.
__global__ __launch_bounds__(256) void fgs_attn(
    const unsigned short* __restrict__ kb, const unsigned short* __restrict__ vb,
    const float* __restrict__ q, float* __restrict__ U, float* __restrict__ Z,
    float* __restrict__ sbuf, int fin) {
  __shared__ float qs[64];
  __shared__ float Uacc[64];
  __shared__ float Zsh;
  __shared__ __attribute__((aligned(16))) unsigned short Kt[2][64][64];
  __shared__ __attribute__((aligned(16))) unsigned short Vt[2][64][64];
  int bid = blockIdx.x;
  int chunk = bid & 7, h = (bid >> 3) & 3, b = bid >> 5;
  int t = threadIdx.x;
  if (t < 64) { qs[t] = q[(size_t)(b * 4 + h) * 64 + t]; Uacc[t] = 0.f; }
  if (t == 0) Zsh = 0.f;
  __syncthreads();

  int g = t >> 2, li = t & 3;         // 4 lanes cooperate per patch n; also row/seg for loads
  float acc[16];
#pragma unroll
  for (int i = 0; i < 16; ++i) acc[i] = 0.f;
  float zacc = 0.f;
  float ql[16];
#pragma unroll
  for (int i = 0; i < 16; ++i) ql[i] = qs[li * 16 + i];

  size_t chunkBase = ((size_t)b * N_ + chunk * 512 + g) * 256 + h * 64 + li * 16;

#if FGS_ASYNC
  {
    size_t gb = chunkBase;            // tile 0
    asyncLoad128(kb + gb,     &Kt[0][g][li * 16]);
    asyncLoad128(kb + gb + 8, &Kt[0][g][li * 16 + 8]);
    asyncLoad128(vb + gb,     &Vt[0][g][li * 16]);
    asyncLoad128(vb + gb + 8, &Vt[0][g][li * 16 + 8]);
  }
#endif
  for (int it = 0; it < 8; ++it) {
    int buf = it & 1;
#if FGS_ASYNC
    if (it < 7) {
      int nb = buf ^ 1;
      size_t gb = chunkBase + (size_t)(it + 1) * 64 * 256;
      asyncLoad128(kb + gb,     &Kt[nb][g][li * 16]);
      asyncLoad128(kb + gb + 8, &Kt[nb][g][li * 16 + 8]);
      asyncLoad128(vb + gb,     &Vt[nb][g][li * 16]);
      asyncLoad128(vb + gb + 8, &Vt[nb][g][li * 16 + 8]);
      __builtin_amdgcn_s_wait_asynccnt(4);   // tile `it` landed (in-order)
    } else {
      __builtin_amdgcn_s_wait_asynccnt(0);
    }
    __syncthreads();
#else
    buf = 0;
    size_t gb = chunkBase + (size_t)it * 64 * 256;
    __builtin_prefetch(kb + gb + 64 * 256, 0, 1);   // global_prefetch_b8
    *reinterpret_cast<uint4*>(&Kt[0][g][li * 16])     = *reinterpret_cast<const uint4*>(kb + gb);
    *reinterpret_cast<uint4*>(&Kt[0][g][li * 16 + 8]) = *reinterpret_cast<const uint4*>(kb + gb + 8);
    *reinterpret_cast<uint4*>(&Vt[0][g][li * 16])     = *reinterpret_cast<const uint4*>(vb + gb);
    *reinterpret_cast<uint4*>(&Vt[0][g][li * 16 + 8]) = *reinterpret_cast<const uint4*>(vb + gb + 8);
    __syncthreads();
#endif
    // compute on tile `it` in buffer `buf`; group g owns patch n
    int n = chunk * 512 + it * 64 + g;
    uint4 k0 = *reinterpret_cast<const uint4*>(&Kt[buf][g][li * 16]);
    uint4 k1 = *reinterpret_cast<const uint4*>(&Kt[buf][g][li * 16 + 8]);
    float kf[16];
    unpack8(k0, kf); unpack8(k1, kf + 8);
    float d = 0.f;
#pragma unroll
    for (int i = 0; i < 16; ++i) d += kf[i] * ql[i];
    d += __shfl_xor(d, 1, 32);
    d += __shfl_xor(d, 2, 32);
    float sg = 1.f / (1.f + expf(-d)) + EPSW;
    if (li == 0) {
      zacc += sg;
      if (fin) sbuf[(size_t)(b * 4 + h) * N_ + n] = sg;
    }
    uint4 v0 = *reinterpret_cast<const uint4*>(&Vt[buf][g][li * 16]);
    uint4 v1 = *reinterpret_cast<const uint4*>(&Vt[buf][g][li * 16 + 8]);
    float vf[16];
    unpack8(v0, vf); unpack8(v1, vf + 8);
#pragma unroll
    for (int i = 0; i < 16; ++i) acc[i] += sg * vf[i];
    __syncthreads();   // done reading `buf` before it is refilled
  }
#pragma unroll
  for (int i = 0; i < 16; ++i) atomicAdd(&Uacc[li * 16 + i], acc[i]);
  if (li == 0) atomicAdd(&Zsh, zacc);
  __syncthreads();
  if (t < 64) atomicAdd(&U[(size_t)(b * 4 + h) * 64 + t], Uacc[t]);
  if (t == 0) atomicAdd(&Z[b * 4 + h], Zsh);
}

// ---- 4: att = U/Z ; GRU ; residual LN+MLP ; rep update ----
__global__ __launch_bounds__(256) void fgs_update(
    const float* __restrict__ U, const float* __restrict__ Z,
    float* __restrict__ rep,
    const float* __restrict__ Wih, const float* __restrict__ Whh,
    const float* __restrict__ bih, const float* __restrict__ bhh,
    const float* __restrict__ lg, const float* __restrict__ lb,
    const float* __restrict__ W1, const float* __restrict__ W2,
    const float* __restrict__ b2, float* __restrict__ out, int fin) {
  __shared__ float att_s[256], rep_s[256], gi_s[768], gh_s[768];
  __shared__ float new_s[256], y_s[256], m1_s[1024];
  __shared__ float red[8];
  int b = blockIdx.x, t = threadIdx.x;
  att_s[t] = U[(size_t)b * 256 + t] / Z[b * 4 + (t >> 6)];
  rep_s[t] = rep[(size_t)b * 256 + t];
  __syncthreads();
#pragma unroll
  for (int jj = 0; jj < 3; ++jj) {
    int j = t + jj * 256;
    float a1 = bih[j], a2 = bhh[j];
    const float* wi = Wih + (size_t)j * 256;
    const float* wh = Whh + (size_t)j * 256;
    for (int d = 0; d < 256; ++d) { a1 += wi[d] * att_s[d]; a2 += wh[d] * rep_s[d]; }
    gi_s[j] = a1; gh_s[j] = a2;
  }
  __syncthreads();
  {
    float r  = 1.f / (1.f + expf(-(gi_s[t] + gh_s[t])));
    float z  = 1.f / (1.f + expf(-(gi_s[256 + t] + gh_s[256 + t])));
    float gg = tanhf(gi_s[512 + t] + r * gh_s[512 + t]);
    new_s[t] = (1.f - z) * gg + z * rep_s[t];
  }
  __syncthreads();
  float nv = new_s[t];
  float s1 = blockSum256(nv, red);
  float s2 = blockSum256(nv * nv, red);
  float mean = s1 * (1.f / 256.f);
  float var = s2 * (1.f / 256.f) - mean * mean;
  y_s[t] = (nv - mean) * rsqrtf(var + 1e-5f) * lg[t] + lb[t];
  __syncthreads();
#pragma unroll
  for (int hh = 0; hh < 4; ++hh) {
    int idx = t + hh * 256;
    float a = 0.f;
    for (int d = 0; d < 256; ++d) a += y_s[d] * W1[(size_t)d * 1024 + idx];
    m1_s[idx] = fmaxf(a, 0.f);
  }
  __syncthreads();
  float o = nv + b2[t];
  for (int m = 0; m < 1024; ++m) o += m1_s[m] * W2[(size_t)m * 256 + t];
  rep[(size_t)b * 256 + t] = o;
  if (fin) out[(size_t)b * 256 + t] = o;
}

// ---- 5: w_out[b,n] = sum_h s[b,h,n] / Z[b,h] ----
__global__ __launch_bounds__(256) void fgs_finw(
    const float* __restrict__ sbuf, const float* __restrict__ Z,
    float* __restrict__ wout) {
  int i = blockIdx.x * 256 + threadIdx.x;   // B*N
  int b = i >> 12, n = i & 4095;
  float acc = 0.f;
#pragma unroll
  for (int h = 0; h < 4; ++h)
    acc += sbuf[(size_t)(b * 4 + h) * N_ + n] / Z[b * 4 + h];
  wout[i] = acc;
}

extern "C" void kernel_launch(void* const* d_in, const int* in_sizes, int n_in,
                              void* d_out, int out_size, void* d_ws, size_t ws_size,
                              hipStream_t stream) {
  (void)in_sizes; (void)n_in; (void)out_size; (void)ws_size;
  const float* inputs  = (const float*)d_in[0];
  const float* eps     = (const float*)d_in[1];
  const float* init_mu = (const float*)d_in[2];
  const float* init_lv = (const float*)d_in[3];
  const float* Wk      = (const float*)d_in[4];
  const float* Wv      = (const float*)d_in[5];
  const float* Wq      = (const float*)d_in[6];
  const float* ln_in_g = (const float*)d_in[7];
  const float* ln_in_b = (const float*)d_in[8];
  const float* ln_sg   = (const float*)d_in[9];
  const float* ln_sb   = (const float*)d_in[10];
  const float* ln_rg   = (const float*)d_in[11];
  const float* ln_rb   = (const float*)d_in[12];
  const float* Wih     = (const float*)d_in[13];
  const float* Whh     = (const float*)d_in[14];
  const float* bih     = (const float*)d_in[15];
  const float* bhh     = (const float*)d_in[16];
  const float* W1      = (const float*)d_in[17];
  const float* W2      = (const float*)d_in[18];
  const float* b2      = (const float*)d_in[19];

  char* ws = (char*)d_ws;
  unsigned short* kbuf = (unsigned short*)(ws + KOFF);
  unsigned short* vbuf = (unsigned short*)(ws + VOFF);
  unsigned short* wT   = (unsigned short*)(ws + WOFF);
  float* qbuf = (float*)(ws + QOFF);
  float* Ubuf = (float*)(ws + UOFF);
  float* Zbuf = (float*)(ws + ZOFF);
  float* sbuf = (float*)(ws + SOFF);
  float* rep  = (float*)(ws + ROFF);
  float* out  = (float*)d_out;

  fgs_prep<<<256, 256, 0, stream>>>(Wk, Wv, init_mu, init_lv, eps, wT, rep);
  fgs_lnproj<<<8192, 256, 0, stream>>>(inputs, ln_in_g, ln_in_b, wT, kbuf, vbuf);
  for (int s = 0; s < 4; ++s) {
    int fin = (s == 3);
    fgs_q<<<128, 256, 0, stream>>>(rep, Wq, ln_sg, ln_sb, qbuf, Ubuf, Zbuf);
    fgs_attn<<<4096, 256, 0, stream>>>(kbuf, vbuf, qbuf, Ubuf, Zbuf, sbuf, fin);
    fgs_update<<<128, 256, 0, stream>>>(Ubuf, Zbuf, rep, Wih, Whh, bih, bhh,
                                        ln_rg, ln_rb, W1, W2, b2, out, fin);
  }
  fgs_finw<<<2048, 256, 0, stream>>>(sbuf, Zbuf, out + B_ * D_);
}
